// DPGGAN_12240656794038
// MI455X (gfx1250) — matrix operand order repaired and checked by reading, not compiled, not run on hardware
//
#include <hip/hip_runtime.h>
#include <hip/hip_bf16.h>
#include <math.h>

// ---------------------------------------------------------------------------
// DP-GGAN forward on gfx1250 (wave32, WMMA bf16 -> f32 accumulate)
// ---------------------------------------------------------------------------

typedef __bf16 bhalf;
typedef __attribute__((ext_vector_type(16))) __bf16 v16bf;
typedef __attribute__((ext_vector_type(8)))  float  v8f;
typedef __attribute__((ext_vector_type(4)))  __bf16 bf4;

#define B_    4096
#define S_    10
#define FEAT_ 256
#define L1_   256
#define L2_   128
#define D2_   256

// ---------------- reductions (wave32) ----------------
__device__ __forceinline__ float wave_sum(float v) {
#pragma unroll
  for (int off = 16; off > 0; off >>= 1) v += __shfl_xor(v, off, 32);
  return v;
}

__device__ __forceinline__ float block_sum(float v) {
  __shared__ float s[32];
  v = wave_sum(v);
  const int lane = threadIdx.x & 31, w = threadIdx.x >> 5;
  if (lane == 0) s[w] = v;
  __syncthreads();
  const int nw = blockDim.x >> 5;
  v = (threadIdx.x < (unsigned)nw) ? s[threadIdx.x] : 0.0f;
  if (w == 0) v = wave_sum(v);
  return v;
}

// ---------------- WMMA fragment loaders (ISA 7.12.2 layouts) ----------------
// A 16x32 bf16: lane holds row M=lane&15; V0-3 = K[sel*8 .. sel*8+7],
//               V4-7 = K[16+sel*8 .. 16+sel*8+7]   (sel = lane>>4)
__device__ __forceinline__ v16bf load_frag_a(const bhalf* __restrict__ p,
                                             int row, int ldk, int k0, int sel) {
  const bhalf* base = p + (size_t)row * ldk + k0 + sel * 8;
  v16bf f;
  ((uint4*)&f)[0] = *(const uint4*)(base);
  ((uint4*)&f)[1] = *(const uint4*)(base + 16);
  return f;
}
// B 32x16 bf16 (stored as Bt[N,K] row-major): lane holds col N=lane&15;
// V0-3 = K[sel*8 .. sel*8+7], V4-7 = K[16+sel*8 .. 16+sel*8+7].
__device__ __forceinline__ v16bf load_frag_b(const bhalf* __restrict__ p,
                                             int col, int ldk, int k0, int sel) {
  const bhalf* base = p + (size_t)col * ldk + k0 + sel * 8;
  v16bf f;
  ((uint4*)&f)[0] = *(const uint4*)(base);
  ((uint4*)&f)[1] = *(const uint4*)(base + 16);
  return f;
}

// ---------------- generic bf16 GEMM: C[M,N] = A[M,K] @ Bt[N,K]^T -------------
// mode 0: store f32 | mode 1: store -f32 | mode 2: relu -> store bf16
__global__ __launch_bounds__(128) void wmma_gemm_bf16(
    const bhalf* __restrict__ A, const bhalf* __restrict__ Bt,
    void* __restrict__ C, int M, int N, int K, int mode) {
  const int lane = threadIdx.x & 31;
  const int wave = threadIdx.x >> 5;
  const int r    = lane & 15;
  const int sel  = lane >> 4;
  const int tileM = blockIdx.y * 64 + (wave >> 1) * 32;
  const int tileN = blockIdx.x * 64 + (wave & 1) * 32;

  v8f acc00 = {}, acc01 = {}, acc10 = {}, acc11 = {};
  for (int k0 = 0; k0 < K; k0 += 32) {
    v16bf a0 = load_frag_a(A, tileM + r,      K, k0, sel);
    v16bf a1 = load_frag_a(A, tileM + 16 + r, K, k0, sel);
    v16bf b0 = load_frag_b(Bt, tileN + r,      K, k0, sel);
    v16bf b1 = load_frag_b(Bt, tileN + 16 + r, K, k0, sel);
    acc00 = __builtin_amdgcn_wmma_f32_16x16x32_bf16(false, a0, false, b0, (short)0, acc00, false, false);
    acc01 = __builtin_amdgcn_wmma_f32_16x16x32_bf16(false, a0, false, b1, (short)0, acc01, false, false);
    acc10 = __builtin_amdgcn_wmma_f32_16x16x32_bf16(false, a1, false, b0, (short)0, acc10, false, false);
    acc11 = __builtin_amdgcn_wmma_f32_16x16x32_bf16(false, a1, false, b1, (short)0, acc11, false, false);
  }

  float* Cf = (float*)C;
  bhalf* Cb = (bhalf*)C;
#pragma unroll
  for (int mi = 0; mi < 2; mi++) {
#pragma unroll
    for (int ni = 0; ni < 2; ni++) {
      v8f acc = (mi == 0) ? (ni == 0 ? acc00 : acc01) : (ni == 0 ? acc10 : acc11);
      const int colx = tileN + ni * 16 + r;
      const size_t rb = (size_t)(tileM + mi * 16 + sel * 8) * N + colx;
#pragma unroll
      for (int rr = 0; rr < 8; rr++) {
        const float v = acc[rr];
        const size_t off = rb + (size_t)rr * N;
        if (mode == 0)      Cf[off] = v;
        else if (mode == 1) Cf[off] = -v;
        else                Cb[off] = (bhalf)(v > 0.0f ? v : 0.0f);
      }
    }
  }
}

// ---------------- small kernels ----------------
// Xagg[n,:] = mean_s feat[neighs2[n,s],:]  -> bf16
__global__ __launch_bounds__(64) void gather_mean_kernel(
    const float* __restrict__ feat, const int* __restrict__ neighs,
    bhalf* __restrict__ out) {
  const int n = blockIdx.x;
  const int t = threadIdx.x;
  const int* idx = neighs + n * S_;
  float ax = 0, ay = 0, az = 0, aw = 0;
#pragma unroll
  for (int s = 0; s < S_; s++) {
    const float4 v = ((const float4*)(feat + (size_t)idx[s] * FEAT_))[t];
    ax += v.x; ay += v.y; az += v.z; aw += v.w;
  }
  const float inv = 1.0f / (float)S_;
  bf4 o;
  o.x = (bhalf)(ax * inv); o.y = (bhalf)(ay * inv);
  o.z = (bhalf)(az * inv); o.w = (bhalf)(aw * inv);
  ((bf4*)(out + (size_t)n * FEAT_))[t] = o;
}

// agg[b,:] = mean_s h1[b*S+s,:]  (bf16 -> bf16, f32 accum)
__global__ __launch_bounds__(64) void neigh_mean_kernel(
    const bhalf* __restrict__ h1, bhalf* __restrict__ agg) {
  const int b = blockIdx.x;
  const int t = threadIdx.x;
  float ax = 0, ay = 0, az = 0, aw = 0;
#pragma unroll
  for (int s = 0; s < S_; s++) {
    bf4 v = ((const bf4*)(h1 + (size_t)(b * S_ + s) * L1_))[t];
    ax += (float)v.x; ay += (float)v.y; az += (float)v.z; aw += (float)v.w;
  }
  const float inv = 1.0f / (float)S_;
  bf4 o;
  o.x = (bhalf)(ax * inv); o.y = (bhalf)(ay * inv);
  o.z = (bhalf)(az * inv); o.w = (bhalf)(aw * inv);
  ((bf4*)(agg + (size_t)b * L1_))[t] = o;
}

// row-wise l2norm (optionally relu after normalize), f32 in -> bf16 out
// 8 waves per block, one wave per row
__global__ __launch_bounds__(256) void l2norm_rows_kernel(
    const float* __restrict__ in, bhalf* __restrict__ out, int ncol, int do_relu) {
  const int lane = threadIdx.x & 31;
  const int wid  = threadIdx.x >> 5;
  const int row  = blockIdx.x * 8 + wid;
  const float* rp = in + (size_t)row * ncol;
  float ss = 0.0f;
  for (int i = lane; i < ncol; i += 32) { const float x = rp[i]; ss += x * x; }
  ss = wave_sum(ss);
  const float inv = 1.0f / fmaxf(sqrtf(ss), 1e-12f);
  bhalf* op = out + (size_t)row * ncol;
  for (int i = lane; i < ncol; i += 32) {
    float x = rp[i] * inv;
    if (do_relu) x = fmaxf(x, 0.0f);
    op[i] = (bhalf)x;
  }
}

// wgd = Wg @ Wdisc ; bgw = dot(bg, Wdisc)
__global__ __launch_bounds__(256) void make_wgd_kernel(
    const float* __restrict__ Wg, const float* __restrict__ Wdisc,
    const float* __restrict__ bg, float* __restrict__ wgd, float* __restrict__ bgw) {
  const int f = threadIdx.x;
  float s = 0.0f;
  for (int j = 0; j < D2_; j++) s += Wg[f * D2_ + j] * Wdisc[j];
  wgd[f] = s;
  if (f == 0) {
    float t = 0.0f;
    for (int j = 0; j < D2_; j++) t += bg[j] * Wdisc[j];
    *bgw = t;
  }
}

// y[b] = dot(feat[nodes[b],:], wgd)
__global__ __launch_bounds__(64) void gather_dot_kernel(
    const float* __restrict__ feat, const int* __restrict__ nodes,
    const float* __restrict__ wgd, float* __restrict__ y) {
  const int b = blockIdx.x;
  const float4 a = ((const float4*)(feat + (size_t)nodes[b] * FEAT_))[threadIdx.x];
  const float4 w = ((const float4*)wgd)[threadIdx.x];
  float s = a.x * w.x + a.y * w.y + a.z * w.z + a.w * w.w;
  s = block_sum(s);
  if (threadIdx.x == 0) y[b] = s;
}

// out[b] = dot(Mtx[b,:], y) + *bgw   (Mtx is [B_, B_] f32)
__global__ __launch_bounds__(256) void matvec_bias_kernel(
    const float* __restrict__ Mtx, const float* __restrict__ y,
    const float* __restrict__ bgw, float* __restrict__ out) {
  const int b = blockIdx.x;
  const float4* rp = (const float4*)(Mtx + (size_t)b * B_);
  const float4* yp = (const float4*)y;
  float s = 0.0f;
  for (int i = threadIdx.x; i < B_ / 4; i += 256) {
    const float4 a = rp[i], w = yp[i];
    s += a.x * w.x + a.y * w.y + a.z * w.z + a.w * w.w;
  }
  s = block_sum(s);
  if (threadIdx.x == 0) out[b] = s + *bgw;
}

__global__ void fill_labels_kernel(float* __restrict__ lab) {
  const int i = blockIdx.x * blockDim.x + threadIdx.x;
  if (i < 2 * B_) lab[i] = (i < B_) ? 1.0f : 0.0f;
}

// f32 [K,N] -> bf16 [N,K] (transpose so GEMM always contracts along rows)
__global__ void convert_transpose_kernel(const float* __restrict__ w,
                                         bhalf* __restrict__ wt, int K, int N) {
  const int i = blockIdx.x * blockDim.x + threadIdx.x;
  if (i < K * N) {
    const int k = i / N, n = i % N;
    wt[(size_t)n * K + k] = (bhalf)w[i];
  }
}

// ---------------------------------------------------------------------------
extern "C" void kernel_launch(void* const* d_in, const int* in_sizes, int n_in,
                              void* d_out, int out_size, void* d_ws, size_t ws_size,
                              hipStream_t stream) {
  const int*   nodes   = (const int*)d_in[0];
  const int*   neighs2 = (const int*)d_in[1];
  const float* sub_adj = (const float*)d_in[2];
  const float* feat    = (const float*)d_in[3];
  const float* W1  = (const float*)d_in[4];
  const float* W2  = (const float*)d_in[5];
  const float* W3  = (const float*)d_in[6];
  const float* Wd1 = (const float*)d_in[7];
  const float* Wd2 = (const float*)d_in[8];
  const float* Wm1 = (const float*)d_in[9];
  const float* Wm2 = (const float*)d_in[10];
  const float* Wg  = (const float*)d_in[11];
  const float* bg  = (const float*)d_in[12];
  const float* Wdisc = (const float*)d_in[13];

  // ---- workspace layout (static, deterministic) ----
  char* ws = (char*)d_ws;
  size_t off = 0;
  auto alloc = [&](size_t bytes) {
    size_t o = off;
    off += (bytes + 255) & ~(size_t)255;
    return (void*)(ws + o);
  };
  bhalf* XAGG = (bhalf*)alloc((size_t)B_ * S_ * FEAT_ * 2);  // 20 MB
  bhalf* H1   = (bhalf*)alloc((size_t)B_ * S_ * L1_ * 2);    // 20 MB
  bhalf* AGG  = (bhalf*)alloc((size_t)B_ * L1_ * 2);         // 2 MB
  bhalf* Hh   = (bhalf*)alloc((size_t)B_ * L2_ * 2);         // 1 MB
  float* TMP  = (float*)alloc((size_t)B_ * D2_ * 4);         // 4 MB
  bhalf* O1   = (bhalf*)alloc((size_t)B_ * D2_ * 2);
  bhalf* O2   = (bhalf*)alloc((size_t)B_ * D2_ * 2);
  bhalf* E1   = (bhalf*)alloc((size_t)B_ * D2_ * 2);
  bhalf* E2   = (bhalf*)alloc((size_t)B_ * D2_ * 2);
  bhalf* W1T  = (bhalf*)alloc((size_t)L1_ * FEAT_ * 2);
  bhalf* W2T  = (bhalf*)alloc((size_t)L2_ * L1_ * 2);
  bhalf* W3T  = (bhalf*)alloc((size_t)L2_ * L1_ * 2);
  bhalf* WD1T = (bhalf*)alloc((size_t)D2_ * L2_ * 2);
  bhalf* WD2T = (bhalf*)alloc((size_t)D2_ * D2_ * 2);
  bhalf* WM1T = (bhalf*)alloc((size_t)D2_ * D2_ * 2);
  bhalf* WM2T = (bhalf*)alloc((size_t)D2_ * D2_ * 2);
  float* WGD  = (float*)alloc(D2_ * 4);
  float* Y    = (float*)alloc(B_ * 4);
  float* BGW  = (float*)alloc(256);

  // ---- output layout (flat concat in return order) ----
  float* out_mu   = (float*)d_out;                       // [B, L2]
  float* out_lv   = out_mu + (size_t)B_ * L2_;           // [B, L2]
  float* out_rc   = out_lv + (size_t)B_ * L2_;           // [B, B]
  float* out_pred = out_rc + (size_t)B_ * B_;            // [2B, 1]
  float* out_lab  = out_pred + 2 * B_;                   // [2B, 1]

  // ---- weight conversion (transpose -> bf16 [N,K]) ----
  auto convT = [&](const float* w, bhalf* wt, int K, int N) {
    const int tot = K * N;
    convert_transpose_kernel<<<(tot + 255) / 256, 256, 0, stream>>>(w, wt, K, N);
  };
  convT(W1,  W1T,  FEAT_, L1_);
  convT(W2,  W2T,  L1_,   L2_);
  convT(W3,  W3T,  L1_,   L2_);
  convT(Wd1, WD1T, L2_,   D2_);
  convT(Wd2, WD2T, D2_,   D2_);
  convT(Wm1, WM1T, D2_,   D2_);
  convT(Wm2, WM2T, D2_,   D2_);

  // ---- encoder ----
  gather_mean_kernel<<<B_ * S_, 64, 0, stream>>>(feat, neighs2, XAGG);
  // H1 = relu(XAGG @ W1) -> bf16   [40960, 256]
  wmma_gemm_bf16<<<dim3(L1_ / 64, (B_ * S_) / 64), 128, 0, stream>>>(
      XAGG, W1T, H1, B_ * S_, L1_, FEAT_, 2);
  neigh_mean_kernel<<<B_, 64, 0, stream>>>(H1, AGG);
  // mu / logvar_sub
  wmma_gemm_bf16<<<dim3(L2_ / 64, B_ / 64), 128, 0, stream>>>(
      AGG, W2T, out_mu, B_, L2_, L1_, 0);
  wmma_gemm_bf16<<<dim3(L2_ / 64, B_ / 64), 128, 0, stream>>>(
      AGG, W3T, out_lv, B_, L2_, L1_, 1);

  // ---- reparam (training path): h = l2norm(mu) ----
  l2norm_rows_kernel<<<B_ / 8, 256, 0, stream>>>(out_mu, Hh, L2_, 0);

  // ---- decoder ----
  wmma_gemm_bf16<<<dim3(D2_ / 64, B_ / 64), 128, 0, stream>>>(
      Hh, WD1T, TMP, B_, D2_, L2_, 0);
  l2norm_rows_kernel<<<B_ / 8, 256, 0, stream>>>(TMP, O1, D2_, 1);
  wmma_gemm_bf16<<<dim3(D2_ / 64, B_ / 64), 128, 0, stream>>>(
      O1, WD2T, TMP, B_, D2_, D2_, 0);
  l2norm_rows_kernel<<<B_ / 8, 256, 0, stream>>>(TMP, O2, D2_, 1);
  wmma_gemm_bf16<<<dim3(D2_ / 64, B_ / 64), 128, 0, stream>>>(
      O2, WM1T, TMP, B_, D2_, D2_, 0);
  l2norm_rows_kernel<<<B_ / 8, 256, 0, stream>>>(TMP, E1, D2_, 0);
  wmma_gemm_bf16<<<dim3(D2_ / 64, B_ / 64), 128, 0, stream>>>(
      O2, WM2T, TMP, B_, D2_, D2_, 0);
  l2norm_rows_kernel<<<B_ / 8, 256, 0, stream>>>(TMP, E2, D2_, 0);

  // reconst_adj = E1 @ E2^T  [4096,4096] f32 -> d_out
  wmma_gemm_bf16<<<dim3(B_ / 64, B_ / 64), 128, 0, stream>>>(
      E1, E2, out_rc, B_, B_, D2_, 0);

  // ---- discriminator, collapsed: (A@xw+bg)@Wdisc = A@(feat[nodes]@(Wg@Wdisc)) + bg.Wdisc
  make_wgd_kernel<<<1, 256, 0, stream>>>(Wg, Wdisc, bg, WGD, BGW);
  gather_dot_kernel<<<B_, 64, 0, stream>>>(feat, nodes, WGD, Y);
  matvec_bias_kernel<<<B_, 256, 0, stream>>>(sub_adj, Y, BGW, out_pred);        // orig_prob
  matvec_bias_kernel<<<B_, 256, 0, stream>>>(out_rc, Y, BGW, out_pred + B_);    // gen_prob
  fill_labels_kernel<<<(2 * B_ + 255) / 256, 256, 0, stream>>>(out_lab);
}
